// EmbResBlock_32796370272501
// MI455X (gfx1250) — matrix operand order repaired and checked by reading, not compiled, not run on hardware
//
#include <hip/hip_runtime.h>
#include <hip/hip_bf16.h>

// ---------------------------------------------------------------------------
// EmbResBlock for MI455X (gfx1250, wave32, WMMA).
//  conv = gather(h1[idx]) @ W[k]  accumulated over 27 taps, C=64 -> Co=64.
//  v_wmma_f32_16x16x32_f16: 16-node tile per wave, K-dim = channels.
//  Weights pre-swizzled into B-operand fragments, LDS-resident (K*8 KB),
//  loaded with async-to-LDS when available. Gather loop software-pipelined.
// ---------------------------------------------------------------------------

typedef _Float16 v16h __attribute__((ext_vector_type(16)));
typedef _Float16 v8h  __attribute__((ext_vector_type(8)));
typedef float    v8f  __attribute__((ext_vector_type(8)));

// Types for the async-to-LDS builtin (param is `int __vector_size__(16)` ptr,
// AS1 for global / AS3 for LDS, per the hipcc diagnostic).
typedef int tdm_i4 __attribute__((vector_size(16)));
typedef __attribute__((address_space(1))) tdm_i4* gptr_i4;
typedef __attribute__((address_space(3))) tdm_i4* lptr_i4;

#define CONV_THREADS 512   // 16 waves -> 4 waves/SIMD even at 1 WG/WGP (221KB LDS)

__device__ __forceinline__ float silu_f(float x) {
    return x / (1.0f + __expf(-x));
}

// ---------------------------------------------------------------------------
// FiLM table: film[b][0:64] = scale, film[b][64:128] = shift
// ---------------------------------------------------------------------------
__global__ void film_kernel(const float* __restrict__ t, const float* __restrict__ Wt,
                            const float* __restrict__ bt, float* __restrict__ film, int Bn) {
    int tid = blockIdx.x * blockDim.x + threadIdx.x;
    if (tid >= Bn * 128) return;
    int bi = tid >> 7, o = tid & 127;
    float s = bt[o];
    for (int e = 0; e < 256; ++e) {
        float tv = t[bi * 256 + e];
        s += silu_f(tv) * Wt[e * 128 + o];
    }
    film[tid] = s;
}

// ---------------------------------------------------------------------------
// Fused eval-BatchNorm + SiLU + f32->f16 convert (feeds the WMMA gather).
// ---------------------------------------------------------------------------
__global__ void bnsilu_cvt_kernel(const float* __restrict__ x,
                                  const float* __restrict__ g, const float* __restrict__ b,
                                  const float* __restrict__ m, const float* __restrict__ v,
                                  _Float16* __restrict__ out, int total) {
    int i = blockIdx.x * blockDim.x + threadIdx.x;
    if (i >= total) return;
    int c = i & 63;
    float y = (x[i] - m[c]) * rsqrtf(v[c] + 1e-5f) * g[c] + b[c];
    out[i] = (_Float16)silu_f(y);
}

// ---------------------------------------------------------------------------
// Swizzle W [K,64,64] f32 -> B-operand fragments, f16.
// Block (k, ctile, ftile) of 512 halfs, lane-major:
//   Ws[blk*512 + lane*16 + e] = W[k][ctile*32 + (e>>3)*16 + (lane>>4)*8 + (e&7)]
//                                [ftile*16 + (lane&15)]
// -> one contiguous 32-byte run per lane (2x ds_load_b128 at use time).
// ---------------------------------------------------------------------------
__global__ void swizzleW_kernel(const float* __restrict__ W, _Float16* __restrict__ Ws, int K) {
    int o = blockIdx.x * blockDim.x + threadIdx.x;
    if (o >= K * 4096) return;
    int blk    = o >> 9;
    int within = o & 511;
    int lane = within >> 4;
    int e    = within & 15;
    int v = e >> 1, lo = e & 1;
    int k  = blk >> 3;
    int ct = (blk >> 2) & 1;
    int ft = blk & 3;
    int c = ct * 32 + (e >> 3) * 16 + (lane >> 4) * 8 + (e & 7);   // B-matrix K index
    int f = ft * 16 + (lane & 15);                     // B-matrix N index
    Ws[o] = (_Float16)W[(k * 64 + c) * 64 + f];
}

// ---------------------------------------------------------------------------
// Sparse conv via WMMA, two-stage software-pipelined gather.
// MODE 0: conv1 epilogue (+b1c, FiLM, BN2+SiLU, f16 store)
// MODE 1: conv2 epilogue (+b2c, +x residual, f32 store)
// ---------------------------------------------------------------------------
template <int MODE>
__global__ void __launch_bounds__(CONV_THREADS)
conv_wmma_kernel(const _Float16* __restrict__ feat,
                 const _Float16* __restrict__ Ws,
                 const int* __restrict__ kidx, const int* __restrict__ kval,
                 const float* __restrict__ bias,
                 const int* __restrict__ bidx, const float* __restrict__ film,
                 const float* __restrict__ bn_g, const float* __restrict__ bn_b,
                 const float* __restrict__ bn_m, const float* __restrict__ bn_v,
                 const float* __restrict__ xres,
                 _Float16* __restrict__ out16, float* __restrict__ out32,
                 int N, int K) {
    extern __shared__ _Float16 ldsW[];

    // Weights -> LDS, once per workgroup. Prefer the CDNA5 async-to-LDS path.
    {
        const int chunks = K * 512;  // 16-byte chunks
#if __has_builtin(__builtin_amdgcn_global_load_async_to_lds_b128)
        for (int i = threadIdx.x; i < chunks; i += blockDim.x) {
            __builtin_amdgcn_global_load_async_to_lds_b128(
                (gptr_i4)(void*)(Ws + (size_t)i * 8),
                (lptr_i4)(void*)(ldsW + (size_t)i * 8),
                0, 0);
        }
#if __has_builtin(__builtin_amdgcn_s_wait_asynccnt)
        __builtin_amdgcn_s_wait_asynccnt(0);
#else
        asm volatile("s_wait_asynccnt 0x0" ::: "memory");
#endif
#else
        const int4* src = (const int4*)Ws;
        int4* dst = (int4*)ldsW;
        for (int i = threadIdx.x; i < chunks; i += blockDim.x) dst[i] = src[i];
#endif
    }
    __syncthreads();

    const int lane  = threadIdx.x & 31;
    const int wave  = threadIdx.x >> 5;
    const int tiles = (N + 15) >> 4;
    const int tile  = blockIdx.x * (blockDim.x >> 5) + wave;
    if (tile >= tiles) return;  // wave-uniform: EXEC stays all-1s for WMMA

    const int khalf = lane >> 4;
    const int n0    = tile << 4;
    int nn = n0 + (lane & 15);
    if (nn >= N) nn = N - 1;

    v8f acc[4] = {v8f{}, v8f{}, v8f{}, v8f{}};

    // --- software pipeline: idx two stages ahead, A-rows one stage ahead ---
    int vcur = kval[0 * N + nn];
    int rcur = kidx[0 * N + nn];
    const int k1 = (K > 1) ? 1 : 0;
    int vnxt = kval[k1 * N + nn];
    int rnxt = kidx[k1 * N + nn];

    const _Float16* ar0 = feat + ((size_t)rcur << 6) + (khalf << 3);
    v8h a00 = *(const v8h*)(ar0);
    v8h a01 = *(const v8h*)(ar0 + 16);
    v8h a10 = *(const v8h*)(ar0 + 32);
    v8h a11 = *(const v8h*)(ar0 + 48);

    for (int k = 0; k < K; ++k) {
        // stage k+2: fetch tap index/valid (clamped; redundant tail loads ok)
        const int k2 = (k + 2 < K) ? (k + 2) : (K - 1);
        const int r2 = kidx[k2 * N + nn];
        const int v2 = kval[k2 * N + nn];

        // stage k+1: gather next A rows (index fetched one iteration ago)
        const _Float16* an = feat + ((size_t)rnxt << 6) + (khalf << 3);
        v8h n00 = *(const v8h*)(an);
        v8h n01 = *(const v8h*)(an + 16);
        v8h n10 = *(const v8h*)(an + 32);
        v8h n11 = *(const v8h*)(an + 48);

        // stage k: consume current A with 8 WMMAs against LDS-resident B
        const v16h zero = {};
        v16h aA = __builtin_shufflevector(a00, a01,
                0, 1, 2, 3, 4, 5, 6, 7, 8, 9, 10, 11, 12, 13, 14, 15);
        v16h aB = __builtin_shufflevector(a10, a11,
                0, 1, 2, 3, 4, 5, 6, 7, 8, 9, 10, 11, 12, 13, 14, 15);
        aA = vcur ? aA : zero;  // invalid tap -> zero row, EXEC untouched
        aB = vcur ? aB : zero;

        const _Float16* ldsk = ldsW + (k << 12) + (lane << 4);
#pragma unroll
        for (int ft = 0; ft < 4; ++ft) {
            v8h b0 = *(const v8h*)(ldsk + (ft << 9));
            v8h b1 = *(const v8h*)(ldsk + (ft << 9) + 8);
            v16h bf = __builtin_shufflevector(b0, b1,
                    0, 1, 2, 3, 4, 5, 6, 7, 8, 9, 10, 11, 12, 13, 14, 15);
            acc[ft] = __builtin_amdgcn_wmma_f32_16x16x32_f16(
                    false, aA, false, bf, (short)0, acc[ft], false, false);
        }
#pragma unroll
        for (int ft = 0; ft < 4; ++ft) {
            v8h b0 = *(const v8h*)(ldsk + 2048 + (ft << 9));
            v8h b1 = *(const v8h*)(ldsk + 2048 + (ft << 9) + 8);
            v16h bf = __builtin_shufflevector(b0, b1,
                    0, 1, 2, 3, 4, 5, 6, 7, 8, 9, 10, 11, 12, 13, 14, 15);
            acc[ft] = __builtin_amdgcn_wmma_f32_16x16x32_f16(
                    false, aB, false, bf, (short)0, acc[ft], false, false);
        }

        // rotate pipeline registers
        a00 = n00; a01 = n01; a10 = n10; a11 = n11;
        vcur = vnxt; vnxt = v2;
        rnxt = r2;
    }

    // D layout: VGPR j, lanes 0-15 -> node row j, lanes 16-31 -> node row 8+j;
    // feature column = (lane&15) + 16*ftile.
    if (MODE == 0) {
        int bb[8];
#pragma unroll
        for (int j = 0; j < 8; ++j) {
            int n = n0 + khalf * 8 + j;
            bb[j] = bidx[n < N ? n : (N - 1)];
        }
#pragma unroll
        for (int ft = 0; ft < 4; ++ft) {
            const int f = (lane & 15) + ft * 16;
            const float bi = bias[f];
            const float g = bn_g[f], be = bn_b[f];
            const float rs = rsqrtf(bn_v[f] + 1e-5f), mm = bn_m[f];
#pragma unroll
            for (int j = 0; j < 8; ++j) {
                int n = n0 + khalf * 8 + j;
                if (n < N) {
                    float h = acc[ft][j] + bi;
                    h = (1.0f + film[bb[j] * 128 + f]) * h + film[bb[j] * 128 + 64 + f];
                    float y = (h - mm) * rs * g + be;
                    out16[(size_t)n * 64 + f] = (_Float16)silu_f(y);
                }
            }
        }
    } else {
#pragma unroll
        for (int ft = 0; ft < 4; ++ft) {
            const int f = (lane & 15) + ft * 16;
            const float bi = bias[f];
#pragma unroll
            for (int j = 0; j < 8; ++j) {
                int n = n0 + khalf * 8 + j;
                if (n < N) {
                    out32[(size_t)n * 64 + f] = acc[ft][j] + bi + xres[(size_t)n * 64 + f];
                }
            }
        }
    }
}

// ---------------------------------------------------------------------------
extern "C" void kernel_launch(void* const* d_in, const int* in_sizes, int n_in,
                              void* d_out, int out_size, void* d_ws, size_t ws_size,
                              hipStream_t stream) {
    const float* x    = (const float*)d_in[0];
    const float* t    = (const float*)d_in[1];
    const int*   b    = (const int*)d_in[2];
    const int*   kidx = (const int*)d_in[3];
    const int*   kval = (const int*)d_in[4];
    const float* bn1g = (const float*)d_in[5];
    const float* bn1b = (const float*)d_in[6];
    const float* bn1m = (const float*)d_in[7];
    const float* bn1v = (const float*)d_in[8];
    const float* W1   = (const float*)d_in[9];
    const float* b1c  = (const float*)d_in[10];
    const float* bn2g = (const float*)d_in[11];
    const float* bn2b = (const float*)d_in[12];
    const float* bn2m = (const float*)d_in[13];
    const float* bn2v = (const float*)d_in[14];
    const float* W2   = (const float*)d_in[15];
    const float* b2c  = (const float*)d_in[16];
    const float* Wt   = (const float*)d_in[17];
    const float* bt   = (const float*)d_in[18];
    float* out = (float*)d_out;

    const int N  = in_sizes[0] / 64;
    const int K  = in_sizes[3] / N;    // 27
    const int Bn = in_sizes[1] / 256;  // 16

    // Workspace layout (256B-aligned slabs)
    char* ws = (char*)d_ws;
    size_t off = 0;
    auto take = [&](size_t bytes) { size_t o = off; off = (off + bytes + 255) & ~(size_t)255; return o; };
    float*    film = (float*)(ws + take((size_t)Bn * 128 * sizeof(float)));
    _Float16* W1s  = (_Float16*)(ws + take((size_t)K * 4096 * sizeof(_Float16)));
    _Float16* W2s  = (_Float16*)(ws + take((size_t)K * 4096 * sizeof(_Float16)));
    _Float16* h1   = (_Float16*)(ws + take((size_t)N * 64 * sizeof(_Float16)));
    _Float16* h2   = (_Float16*)(ws + take((size_t)N * 64 * sizeof(_Float16)));
    (void)ws_size; (void)n_in; (void)out_size;

    // 1) FiLM table
    film_kernel<<<(Bn * 128 + 255) / 256, 256, 0, stream>>>(t, Wt, bt, film, Bn);
    // 2) BN1 + SiLU + f16 convert
    bnsilu_cvt_kernel<<<((N * 64) + 255) / 256, 256, 0, stream>>>(x, bn1g, bn1b, bn1m, bn1v, h1, N * 64);
    // 3) Weight swizzles for the WMMA B-operand layout
    swizzleW_kernel<<<(K * 4096 + 255) / 256, 256, 0, stream>>>(W1, W1s, K);
    swizzleW_kernel<<<(K * 4096 + 255) / 256, 256, 0, stream>>>(W2, W2s, K);

    const int tiles      = (N + 15) / 16;
    const int wavesPerWg = CONV_THREADS / 32;
    const int wgs        = (tiles + wavesPerWg - 1) / wavesPerWg;
    const size_t smem    = (size_t)K * 8192;   // swizzled weights in LDS (221 KB)

    // 4) conv1 + FiLM + BN2/SiLU -> h2 (f16)
    conv_wmma_kernel<0><<<wgs, CONV_THREADS, smem, stream>>>(h1, W1s, kidx, kval, b1c,
                                                             b, film, bn2g, bn2b, bn2m, bn2v,
                                                             nullptr, h2, nullptr, N, K);
    // 5) conv2 + residual -> out (f32)
    conv_wmma_kernel<1><<<wgs, CONV_THREADS, smem, stream>>>(h2, W2s, kidx, kval, b2c,
                                                             nullptr, nullptr, nullptr, nullptr, nullptr, nullptr,
                                                             x, nullptr, out, N, K);
}
